// MultiHeadSelfAttention_867583393924
// MI455X (gfx1250) — matrix operand order repaired and checked by reading, not compiled, not run on hardware
//
#include <hip/hip_runtime.h>
#include <hip/hip_bf16.h>

// ---------------- CDNA5 WMMA types ----------------
typedef __attribute__((ext_vector_type(16))) __bf16 v16bf;
typedef __attribute__((ext_vector_type(8)))  float  v8f;
typedef int v4i __attribute__((vector_size(16)));   // matches async-LDS builtin param

union Frag {                 // one 16x32 bf16 WMMA operand (A or B), 8 VGPRs
    v16bf          v;
    unsigned short u[16];
    uint4          q[2];
};

__device__ __forceinline__ unsigned short f2bf(float f) {
    unsigned int x = __float_as_uint(f);
    x += 0x7FFFu + ((x >> 16) & 1u);   // round-to-nearest-even
    return (unsigned short)(x >> 16);
}

__device__ __forceinline__ v8f wmma_bf16(const Frag& a, const Frag& b, v8f c) {
    return __builtin_amdgcn_wmma_f32_16x16x32_bf16(
        false, a.v, false, b.v, (short)0, c, false, false);
}

// ---- CDNA5 async global->LDS copy (ASYNCcnt), with safe fallback ----
__device__ __forceinline__ void async_copy_b128(const unsigned short* g, unsigned short* l) {
#if __has_builtin(__builtin_amdgcn_global_load_async_to_lds_b128)
    __builtin_amdgcn_global_load_async_to_lds_b128(
        (__attribute__((address_space(1))) v4i*)(uintptr_t)g,
        (__attribute__((address_space(3))) v4i*)(unsigned)(uintptr_t)l,
        0, 0);
#else
    *(uint4*)l = *(const uint4*)g;   // synchronous fallback
#endif
}
__device__ __forceinline__ void wait_async() {
#if __has_builtin(__builtin_amdgcn_s_wait_asynccnt)
    __builtin_amdgcn_s_wait_asynccnt(0);
#elif __has_builtin(__builtin_amdgcn_global_load_async_to_lds_b128)
    asm volatile("s_wait_asynccnt 0x0" ::: "memory");
#endif
}

#define NTOK  8192   // B*S
#define EMB   1024
#define SEQ   2048
#define NH    16
#define HD    64
#define PAD   72     // ushorts per staged LDS row (64 data + 8 pad, 144B stride)

// =====================================================================
// One-shot fp32 -> bf16 conversion (bandwidth bound), 8 elems/thread
// =====================================================================
__global__ __launch_bounds__(256)
void cvt_bf16(const float* __restrict__ in, unsigned short* __restrict__ out, int n8) {
    const int i = blockIdx.x * 256 + threadIdx.x;
    if (i >= n8) return;
    const float4* p = (const float4*)in + 2 * (size_t)i;
    const float4 f0 = p[0], f1 = p[1];
    uint4 o;
    o.x = (unsigned)f2bf(f0.x) | ((unsigned)f2bf(f0.y) << 16);
    o.y = (unsigned)f2bf(f0.z) | ((unsigned)f2bf(f0.w) << 16);
    o.z = (unsigned)f2bf(f1.x) | ((unsigned)f2bf(f1.y) << 16);
    o.w = (unsigned)f2bf(f1.z) | ((unsigned)f2bf(f1.w) << 16);
    ((uint4*)out)[i] = o;
}

// =====================================================================
// GEMM:  Out[M,N] = (A[M,K] * B[N,K]^T + bias[N]) * oscale
// A, B both bf16, K-contiguous. Pure b128 loads + WMMA in the hot loop.
// OMODE: 0 = bf16 out scattered as [B,H,S,D]
//        1 = bf16 out scattered as [B,H,D,S]  (V transposed for P*V)
//        2 = fp32 out row-major [M,N]
// block = 128 threads (4 waves), block tile 128x128, wave tile 64x64
// =====================================================================
template<int OMODE>
__global__ __launch_bounds__(128)
void gemm_wmma(const unsigned short* __restrict__ Abf,
               const unsigned short* __restrict__ Bbf,
               const float* __restrict__ bias, void* __restrict__ Out,
               int M, int N, int Kd, float oscale) {
    const int lane = threadIdx.x & 31;
    const int wave = threadIdx.x >> 5;
    const int m    = lane & 15;      // M (A) or N (B) index within fragment
    const int kh   = lane >> 4;      // K-half select
    const int m0   = blockIdx.x * 128 + (wave >> 1) * 64;
    const int n0   = blockIdx.y * 128 + (wave & 1) * 64;

    v8f acc[4][4];
    #pragma unroll
    for (int i = 0; i < 4; ++i)
        #pragma unroll
        for (int j = 0; j < 4; ++j)
            acc[i][j] = (v8f){0,0,0,0,0,0,0,0};

    for (int k0 = 0; k0 < Kd; k0 += 32) {
        Frag a[4];
        #pragma unroll
        for (int i = 0; i < 4; ++i) {
            const unsigned short* rp = Abf + (size_t)(m0 + i * 16 + m) * Kd + k0;
            a[i].q[0] = *(const uint4*)(rp + kh * 8);
            a[i].q[1] = *(const uint4*)(rp + 16 + kh * 8);
        }
        #pragma unroll
        for (int j = 0; j < 4; ++j) {
            Frag b;
            const unsigned short* rp = Bbf + (size_t)(n0 + j * 16 + m) * Kd + k0 + kh * 8;
            b.q[0] = *(const uint4*)(rp);
            b.q[1] = *(const uint4*)(rp + 16);
            #pragma unroll
            for (int i = 0; i < 4; ++i)
                acc[i][j] = wmma_bf16(a[i], b, acc[i][j]);
        }
    }

    // ---- epilogue ----
    #pragma unroll
    for (int j = 0; j < 4; ++j) {
        const int gn = n0 + j * 16 + m;
        const float bv = bias[gn];
        #pragma unroll
        for (int i = 0; i < 4; ++i) {
            #pragma unroll
            for (int r = 0; r < 8; ++r) {
                const int gm = m0 + i * 16 + r + 8 * kh;
                const float v = (acc[i][j][r] + bv) * oscale;
                if (OMODE == 2) {
                    ((float*)Out)[(size_t)gm * N + gn] = v;
                } else {
                    const int bb = gm >> 11, ss = gm & 2047;   // S = 2048
                    const int hh = gn >> 6,  dd = gn & 63;     // D = 64
                    size_t idx;
                    if (OMODE == 0)  // [B,H,S,D]
                        idx = ((size_t)((bb * NH + hh) * SEQ + ss)) * HD + dd;
                    else             // [B,H,D,S]
                        idx = ((size_t)((bb * NH + hh) * HD + dd)) * SEQ + ss;
                    ((unsigned short*)Out)[idx] = f2bf(v);
                }
            }
        }
    }
}

// =====================================================================
// Flash attention: grid (S/64, B*H), block 128 (4 waves, 16 q-rows each)
// K/V 64-key tiles staged once per block in double-buffered LDS via
// async global->LDS copies; all 4 waves consume via ds_load_b128.
// Q pre-scaled by 1/8 in projection.
// Q,K bf16 [B,H,S,D]; Vt bf16 [B,H,D,S]; out bf16 attn [B*S, E]
// =====================================================================
__global__ __launch_bounds__(128)
void attn_flash(const unsigned short* __restrict__ Qb,
                const unsigned short* __restrict__ Kb,
                const unsigned short* __restrict__ Vt,
                const int* __restrict__ maskp,
                unsigned short* __restrict__ attn) {
    __shared__ unsigned short ldsK[2][64 * PAD];       // 2 x 9216 B
    __shared__ unsigned short ldsV[2][64 * PAD];       // 2 x 9216 B
    __shared__ unsigned short plds_all[4 * 16 * PAD];  // per-wave P tiles
    const int tid  = threadIdx.x;
    const int lane = tid & 31;
    const int wave = tid >> 5;
    const int m    = lane & 15;
    const int kh   = lane >> 4;
    const int bh   = blockIdx.y;
    const int b    = bh >> 4, h = bh & 15;
    const int q0   = blockIdx.x * 64 + wave * 16;
    unsigned short* plds = plds_all + wave * 16 * PAD;

    const int* mrow = maskp + b * SEQ;
    const unsigned short* Kbase = Kb + (size_t)bh * SEQ * HD;   // rows contiguous
    const unsigned short* Vbase = Vt + (size_t)bh * HD * SEQ;   // rows stride SEQ

    // issue one 64x64 bf16 tile copy (8 KB): 128 threads x 4 b128 chunks each
    auto stage_tiles = [&](int kt, int buf) {
        #pragma unroll
        for (int i = 0; i < 4; ++i) {
            const int c   = tid + 128 * i;        // chunk 0..511
            const int row = c >> 3, col = c & 7;  // 8 x 16B chunks per row
            // K tile: 8KB contiguous at Kbase + kt*HD
            async_copy_b128(Kbase + (size_t)kt * HD + (size_t)c * 8,
                            &ldsK[buf][row * PAD + col * 8]);
            // V tile: 64 rows (d) of 128B, row stride SEQ elems
            async_copy_b128(Vbase + (size_t)row * SEQ + kt + col * 8,
                            &ldsV[buf][row * PAD + col * 8]);
        }
    };

    // Q fragments (16 rows x 64 d -> two 16x32 A operands), loaded once
    Frag qa[2];
    {
        const unsigned short* qrow = Qb + ((size_t)bh * SEQ + q0 + m) * HD;
        #pragma unroll
        for (int dc = 0; dc < 2; ++dc) {
            qa[dc].q[0] = *(const uint4*)(qrow + dc * 32 + kh * 8);
            qa[dc].q[1] = *(const uint4*)(qrow + dc * 32 + 16 + kh * 8);
        }
    }

    v8f o[4];
    #pragma unroll
    for (int j = 0; j < 4; ++j) o[j] = (v8f){0,0,0,0,0,0,0,0};
    float rmax[8], rsum[8];
    #pragma unroll
    for (int r = 0; r < 8; ++r) { rmax[r] = -1e30f; rsum[r] = 0.f; }

    stage_tiles(0, 0);                       // prefetch tile 0

    for (int t = 0; t < SEQ / 64; ++t) {
        const int cur = t & 1;
        const int kt  = t * 64;
        wait_async();
        __syncthreads();                     // staged tile visible to all waves
        if (t + 1 < SEQ / 64)                // overlap next copy with compute
            stage_tiles(kt + 64, cur ^ 1);

        const unsigned short* kbuf = &ldsK[cur][0];
        const unsigned short* vbuf = &ldsV[cur][0];

        // -------- scores (Q already carries the 1/8 scale) --------
        v8f s[4];
        int mv[4];
        #pragma unroll
        for (int kf = 0; kf < 4; ++kf) {
            s[kf] = (v8f){0,0,0,0,0,0,0,0};
            #pragma unroll
            for (int dc = 0; dc < 2; ++dc) {
                Frag bk;
                const unsigned short* rp = kbuf + (kf * 16 + m) * PAD + dc * 32 + kh * 8;
                bk.q[0] = *(const uint4*)(rp);
                bk.q[1] = *(const uint4*)(rp + 16);
                s[kf] = wmma_bf16(qa[dc], bk, s[kf]);
            }
            mv[kf] = mrow[kt + kf * 16 + m];   // key-padding mask (per column)
        }
        #pragma unroll
        for (int kf = 0; kf < 4; ++kf)
            #pragma unroll
            for (int r = 0; r < 8; ++r)
                s[kf][r] = mv[kf] ? s[kf][r] : -1e30f;

        // -------- online softmax (rows live in 16-lane halves) --------
        #pragma unroll
        for (int r = 0; r < 8; ++r) {
            float vmax = fmaxf(fmaxf(s[0][r], s[1][r]), fmaxf(s[2][r], s[3][r]));
            #pragma unroll
            for (int off = 8; off; off >>= 1)
                vmax = fmaxf(vmax, __shfl_xor(vmax, off));
            const float mn   = fmaxf(rmax[r], vmax);
            const float corr = __expf(rmax[r] - mn);
            rmax[r] = mn;
            float ps = 0.f;
            #pragma unroll
            for (int kf = 0; kf < 4; ++kf) {
                const float p = (s[kf][r] <= -1e29f) ? 0.f : __expf(s[kf][r] - mn);
                s[kf][r] = p;
                ps += p;
            }
            #pragma unroll
            for (int off = 8; off; off >>= 1)
                ps += __shfl_xor(ps, off);
            rsum[r] = rsum[r] * corr + ps;
            #pragma unroll
            for (int j = 0; j < 4; ++j) o[j][r] *= corr;
        }

        // -------- relayout P (C-frag) -> A-frags via per-wave LDS --------
        #pragma unroll
        for (int kf = 0; kf < 4; ++kf)
            #pragma unroll
            for (int r = 0; r < 8; ++r)
                plds[(r + 8 * kh) * PAD + kf * 16 + m] = f2bf(s[kf][r]);
        asm volatile("s_wait_dscnt 0x0" ::: "memory");

        Frag pa[2];
        {
            const unsigned short* pr = plds + m * PAD;
            #pragma unroll
            for (int kc = 0; kc < 2; ++kc) {
                pa[kc].q[0] = *(const uint4*)(pr + kc * 32 + kh * 8);
                pa[kc].q[1] = *(const uint4*)(pr + kc * 32 + 16 + kh * 8);
            }
        }

        // -------- O += P * V (V tile staged transposed: key-contiguous) --------
        #pragma unroll
        for (int j = 0; j < 4; ++j) {
            #pragma unroll
            for (int kc = 0; kc < 2; ++kc) {
                Frag bv;
                const unsigned short* rp = vbuf + (j * 16 + m) * PAD + kc * 32 + kh * 8;
                bv.q[0] = *(const uint4*)(rp);
                bv.q[1] = *(const uint4*)(rp + 16);
                o[j] = wmma_bf16(pa[kc], bv, o[j]);
            }
        }
    }

    // -------- normalize and store bf16 attn [tok, E] --------
    float inv[8];
    #pragma unroll
    for (int r = 0; r < 8; ++r) inv[r] = 1.f / rsum[r];
    #pragma unroll
    for (int j = 0; j < 4; ++j) {
        const int col = h * HD + j * 16 + m;
        #pragma unroll
        for (int r = 0; r < 8; ++r) {
            const int qrow = q0 + r + 8 * kh;
            attn[(size_t)(b * SEQ + qrow) * EMB + col] = f2bf(o[j][r] * inv[r]);
        }
    }
}

// =====================================================================
extern "C" void kernel_launch(void* const* d_in, const int* in_sizes, int n_in,
                              void* d_out, int out_size, void* d_ws, size_t ws_size,
                              hipStream_t stream) {
    (void)in_sizes; (void)n_in; (void)out_size; (void)ws_size;
    const float* x    = (const float*)d_in[0];
    const int*   mask = (const int*)  d_in[1];
    const float* Wq   = (const float*)d_in[2];
    const float* bq   = (const float*)d_in[3];
    const float* Wk   = (const float*)d_in[4];
    const float* bk   = (const float*)d_in[5];
    const float* Wv   = (const float*)d_in[6];
    const float* bv   = (const float*)d_in[7];
    const float* Wo   = (const float*)d_in[8];
    const float* bo   = (const float*)d_in[9];

    // workspace layout (ushort elements): 4x16MB activations + 16MB x + 4x2MB W
    unsigned short* Qb  = (unsigned short*)d_ws;
    unsigned short* Kb  = Qb  + (size_t)NTOK * EMB;
    unsigned short* Vt  = Kb  + (size_t)NTOK * EMB;
    unsigned short* At  = Vt  + (size_t)NTOK * EMB;
    unsigned short* xb  = At  + (size_t)NTOK * EMB;
    unsigned short* wqb = xb  + (size_t)NTOK * EMB;
    unsigned short* wkb = wqb + (size_t)EMB * EMB;
    unsigned short* wvb = wkb + (size_t)EMB * EMB;
    unsigned short* wob = wvb + (size_t)EMB * EMB;

    // ---- one-shot bf16 conversion (x + all four weight matrices) ----
    cvt_bf16<<<(NTOK * EMB / 8 + 255) / 256, 256, 0, stream>>>(x,  xb,  NTOK * EMB / 8);
    cvt_bf16<<<(EMB * EMB / 8 + 255) / 256, 256, 0, stream>>>(Wq, wqb, EMB * EMB / 8);
    cvt_bf16<<<(EMB * EMB / 8 + 255) / 256, 256, 0, stream>>>(Wk, wkb, EMB * EMB / 8);
    cvt_bf16<<<(EMB * EMB / 8 + 255) / 256, 256, 0, stream>>>(Wv, wvb, EMB * EMB / 8);
    cvt_bf16<<<(EMB * EMB / 8 + 255) / 256, 256, 0, stream>>>(Wo, wob, EMB * EMB / 8);

    dim3 ggrid(NTOK / 128, EMB / 128);      // (64, 8)
    // Q projection carries the 1/sqrt(HD) = 1/8 attention scale (exact in bf16)
    gemm_wmma<0><<<ggrid, 128, 0, stream>>>(xb, wqb, bq, Qb, NTOK, EMB, EMB, 0.125f);
    gemm_wmma<0><<<ggrid, 128, 0, stream>>>(xb, wkb, bk, Kb, NTOK, EMB, EMB, 1.0f);
    gemm_wmma<1><<<ggrid, 128, 0, stream>>>(xb, wvb, bv, Vt, NTOK, EMB, EMB, 1.0f);

    attn_flash<<<dim3(SEQ / 64, 4 * NH), 128, 0, stream>>>(Qb, Kb, Vt, mask, At);

    gemm_wmma<2><<<ggrid, 128, 0, stream>>>(At, wob, bo, d_out, NTOK, EMB, EMB, 1.0f);
}
